// ContrastiveLoss_OT_38448547234379
// MI455X (gfx1250) — matrix operand order repaired and checked by reading, not compiled, not run on hardware
//
#include <hip/hip_runtime.h>
#include <stdint.h>

// ---------------------------------------------------------------------------
// Types for CDNA5 WMMA (wave32): bf16 16x16x32, f32 accumulate
// ---------------------------------------------------------------------------
typedef __attribute__((ext_vector_type(16))) __bf16        v16bf;
typedef __attribute__((ext_vector_type(8)))  float         v8f;
typedef __attribute__((ext_vector_type(8)))  unsigned short v8us;

struct TwoV8 { v8us lo, hi; };

// Build a 16-element bf16 fragment from two 16B chunks (LDS).
static __device__ __forceinline__ v16bf frag_from(const unsigned short* p0,
                                                  const unsigned short* p1) {
  TwoV8 t;
  t.lo = *(const v8us*)p0;
  t.hi = *(const v8us*)p1;
  return __builtin_bit_cast(v16bf, t);
}

// f32 -> bf16 round-to-nearest-even (bit trick; NaN edge irrelevant here)
static __device__ __forceinline__ unsigned short f2bf(float f) {
  unsigned u = __float_as_uint(f);
  unsigned r = u + 0x7FFFu + ((u >> 16) & 1u);
  return (unsigned short)(r >> 16);
}

// Problem constants
#define NB   64
#define NR   36
#define NW   32
#define ND   1024

// ---------------------------------------------------------------------------
// Kernel 1: per-row norms + bf16 conversion for im / s / x (xn pre-normalized)
// rows: 0..2303 = im rows, 2304..4351 = s rows, 4352..6399 = x rows
// ---------------------------------------------------------------------------
__global__ void prep_kernel(const float* __restrict__ im, const float* __restrict__ s,
                            const float* __restrict__ x,
                            unsigned short* __restrict__ im_bf,
                            unsigned short* __restrict__ s_bf,
                            unsigned short* __restrict__ xn_bf,
                            float* __restrict__ inv_imn, float* __restrict__ s_norm) {
  __shared__ float red[256];
  const int rid = blockIdx.x, tid = threadIdx.x;
  const float* src; unsigned short* dst; int mode, sub;
  if (rid < 2304)      { sub = rid;        src = im + (size_t)sub * ND; dst = im_bf + (size_t)sub * ND; mode = 0; }
  else if (rid < 4352) { sub = rid - 2304; src = s  + (size_t)sub * ND; dst = s_bf  + (size_t)sub * ND; mode = 1; }
  else                 { sub = rid - 4352; src = x  + (size_t)sub * ND; dst = xn_bf + (size_t)sub * ND; mode = 2; }

  float ss = 0.f;
  for (int e = tid; e < ND; e += 256) { float v = src[e]; ss += v * v; }
  red[tid] = ss; __syncthreads();
  for (int st = 128; st > 0; st >>= 1) { if (tid < st) red[tid] += red[tid + st]; __syncthreads(); }
  const float norm  = sqrtf(red[0]);
  const float scale = (mode == 2) ? 1.f / (norm + 1e-12f) : 1.f;
  for (int e = tid; e < ND; e += 256) dst[e] = f2bf(src[e] * scale);
  if (tid == 0) {
    if (mode == 0)      inv_imn[sub] = 1.f / (norm + 1e-12f);
    else if (mode == 1) s_norm[sub]  = norm;
  }
}

// ---------------------------------------------------------------------------
// Kernel 2: per-image Gram matrix G_i = im[i] @ im[i]^T (36x36, WMMA bf16)
// 9 waves = 288 threads; tile (mt,nt) in 3x3 of 16x16, K = 1024
// Pad rows (36..47) zeroed once; per-chunk loads touch real rows only.
// ---------------------------------------------------------------------------
__global__ void gram_kernel(const unsigned short* __restrict__ im_bf,
                            float* __restrict__ G) {
  __shared__ alignas(16) unsigned short Ab[48 * 64];
  const int i = blockIdx.x, tid = threadIdx.x;
  const int wave = tid >> 5, lane = tid & 31;
  const int mt = wave / 3, nt = wave % 3;
  const int h = lane >> 4, lm = lane & 15;
  const unsigned short* base = im_bf + (size_t)i * NR * ND;

  // zero pad rows once (12 rows * 64 cols = 96 uint4 units)
  if (tid < 96) {
    int row = NR + (tid >> 3), col = (tid & 7) * 8;
    *(uint4*)(Ab + row * 64 + col) = make_uint4(0u, 0u, 0u, 0u);
  }
  const int arow = tid >> 3, acol = (tid & 7) * 8;   // units 0..287 cover rows 0..35

  v8f acc = {};
  for (int kc = 0; kc < 16; ++kc) {
    const int kb = kc * 64;
    if (arow < NR)
      *(uint4*)(Ab + arow * 64 + acol) =
          *(const uint4*)(base + (size_t)arow * ND + kb + acol);
    __syncthreads();
    const unsigned short* ar = Ab + (mt * 16 + lm) * 64;
    const unsigned short* br = Ab + (nt * 16 + lm) * 64;
#pragma unroll
    for (int ks = 0; ks < 2; ++ks) {
      const int k0 = ks * 32;
      v16bf a = frag_from(ar + k0 + h * 8,  ar + k0 + h * 8 + 16);
      v16bf b = frag_from(br + k0 + h * 8,  br + k0 + h * 8 + 16);
      acc = __builtin_amdgcn_wmma_f32_16x16x32_bf16(false, a, false, b, (short)0, acc, false, false);
    }
    __syncthreads();
  }
#pragma unroll
  for (int v = 0; v < 8; ++v) {
    int mr = mt * 16 + v + h * 8;
    int nc = nt * 16 + lm;
    if (mr < NR && nc < NR) G[(size_t)i * (NR * NR) + mr * NR + nc] = acc[v];
  }
}

// ---------------------------------------------------------------------------
// Kernel 3: per-(c,i) pair. Fused, software-pipelined WMMA GEMM:
//   F(48x64) = im[i](48x1024, rows>=36 zero) @ [s[c]; xn[c]]^T (1024x64)
//   raw[r][w]  = F[r][w]      (attention path)
//   raw2[r][w] = F[r][32+w]   (OT cosine path, scaled by inv_imnorm[r])
// Double-buffered LDS staging, register-staged global loads overlapping WMMA,
// one barrier per K-chunk, pad rows zeroed once. 12 waves = 384 threads.
// ---------------------------------------------------------------------------
__global__ void __launch_bounds__(384)
pair_kernel(const unsigned short* __restrict__ im_bf,
            const unsigned short* __restrict__ s_bf,
            const unsigned short* __restrict__ xn_bf,
            const float* __restrict__ inv_imn,
            const float* __restrict__ s_norm,
            const float* __restrict__ G,
            float* __restrict__ scores, float* __restrict__ scores_ot) {
  // staging: A buffers [2][48*64], B buffers [2][64*64] (bf16) = 28 KB
  __shared__ alignas(16) unsigned short sAB[2 * 48 * 64 + 2 * 64 * 64];
  __shared__ float attnS[NW * NR], CS[NW * NR], ES[NW * NR], TS[NW * NR], QS[NW * NR];
  __shared__ float GS[NR * NR];
  __shared__ float rnorm[NR], numS[NW], dlt[NW], sgm[NR], rsim[NW];
  __shared__ float red[512];

  float* F = (float*)sAB;   // 48*64 f32 = 12288 B, aliases the two A buffers exactly

  const int tid  = threadIdx.x;
  const int wave = tid >> 5, lane = tid & 31;
  const int h = lane >> 4, lm = lane & 15;
  const int i = blockIdx.x & 63;     // image index
  const int c = blockIdx.x >> 6;     // caption index
  const int mt = wave >> 2, nt = wave & 3;

  const unsigned short* imrow = im_bf + (size_t)i * NR * ND;
  const unsigned short* srow  = s_bf  + (size_t)c * NW * ND;
  const unsigned short* xrow  = xn_bf + (size_t)c * NW * ND;

  // zero A pad rows (36..47) of BOTH buffers once: 2 * 96 uint4 units
  if (tid < 192) {
    int p = tid / 96, u = tid % 96;
    int row = NR + (u >> 3), col = (u & 7) * 8;
    *(uint4*)(sAB + p * (48 * 64) + row * 64 + col) = make_uint4(0u, 0u, 0u, 0u);
  }

  // per-thread staging registers for the next chunk
  uint4 ra, rb0, rb1;
  const int arow = tid >> 3, acol = (tid & 7) * 8;           // A unit (rows 0..47; >=36 skipped)
  const int b0r  = tid >> 3, b0c  = (tid & 7) * 8;           // B unit tid
  const int b1u  = 384 + tid;
  const int b1r  = b1u >> 3, b1c = (b1u & 7) * 8;            // B unit 384+tid (tid<128)

  auto issue_loads = [&](int kc) {
    const int kb = kc * 64;
    if (arow < NR) ra = *(const uint4*)(imrow + (size_t)arow * ND + kb + acol);
    rb0 = (b0r < NW) ? *(const uint4*)(srow + (size_t)b0r * ND + kb + b0c)
                     : *(const uint4*)(xrow + (size_t)(b0r - NW) * ND + kb + b0c);
    if (tid < 128)
      rb1 = *(const uint4*)(xrow + (size_t)(b1r - NW) * ND + kb + b1c);
  };
  auto store_chunk = [&](int p) {
    unsigned short* Ab = sAB + p * (48 * 64);
    unsigned short* Bb = sAB + 2 * (48 * 64) + p * (64 * 64);
    if (arow < NR) *(uint4*)(Ab + arow * 64 + acol) = ra;
    *(uint4*)(Bb + b0r * 64 + b0c) = rb0;
    if (tid < 128) *(uint4*)(Bb + b1r * 64 + b1c) = rb1;
  };

  // ---- software-pipelined GEMM over K=1024 in 16 chunks of 64 ----
  issue_loads(0);
  store_chunk(0);
  v8f acc = {};
  for (int kc = 0; kc < 16; ++kc) {
    const int p = kc & 1;
    __syncthreads();                         // chunk kc resident in buffer p
    if (kc < 15) issue_loads(kc + 1);        // global loads in flight during WMMA
    const unsigned short* Ab = sAB + p * (48 * 64);
    const unsigned short* Bb = sAB + 2 * (48 * 64) + p * (64 * 64);
    const unsigned short* ar = Ab + (mt * 16 + lm) * 64;
    const unsigned short* br = Bb + (nt * 16 + lm) * 64;
#pragma unroll
    for (int ks = 0; ks < 2; ++ks) {
      const int k0 = ks * 32;
      v16bf a = frag_from(ar + k0 + h * 8,  ar + k0 + h * 8 + 16);
      v16bf b = frag_from(br + k0 + h * 8,  br + k0 + h * 8 + 16);
      acc = __builtin_amdgcn_wmma_f32_16x16x32_bf16(false, a, false, b, (short)0, acc, false, false);
    }
    if (kc < 15) store_chunk(p ^ 1);         // stage next chunk into other buffer
  }
  __syncthreads();

  // spill accumulators into F (aliases A buffers; all WMMA reads completed)
#pragma unroll
  for (int v = 0; v < 8; ++v)
    F[(mt * 16 + v + h * 8) * 64 + nt * 16 + lm] = acc[v];
  __syncthreads();

  // ---- epilogue: stage G, build OT cost C / kernel E / T0, row l2 norms ----
  for (int idx = tid; idx < NR * NR; idx += 384) GS[idx] = G[(size_t)i * (NR * NR) + idx];
  for (int idx = tid; idx < NW * NR; idx += 384) {
    int w = idx / NR, r = idx - w * NR;
    float cv = 1.f - F[r * 64 + NW + w] * inv_imn[i * NR + r];  // cosine distance
    CS[idx] = cv;
    ES[idx] = __expf(-2.f * cv);                                // A = exp(-C/beta), beta=0.5
    TS[idx] = 1.f;
  }
  if (tid < NR) {
    float ssq = 0.f;
    for (int w = 0; w < NW; ++w) { float v = F[tid * 64 + w]; v = v > 0.f ? v : 0.1f * v; ssq += v * v; }
    rnorm[tid] = sqrtf(ssq) + 1e-8f;  // l2norm denom over word axis
    sgm[tid]   = 1.f / (float)NR;     // sigma0
  }
  __syncthreads();

  // attn pre-softmax: (leaky_relu(raw)/rnorm) * lambda_softmax, transposed (w,r)
  for (int idx = tid; idx < NW * NR; idx += 384) {
    int w = idx / NR, r = idx - w * NR;
    float v = F[r * 64 + w]; v = v > 0.f ? v : 0.1f * v;
    attnS[idx] = (v / rnorm[r]) * 9.0f;
  }
  __syncthreads();

  // softmax over regions + num_w = sum_r attn[w,r]*raw[r,w]
  if (tid < NW) {
    float mx = -3.4e38f;
    for (int r = 0; r < NR; ++r) mx = fmaxf(mx, attnS[tid * NR + r]);
    float sm = 0.f;
    for (int r = 0; r < NR; ++r) { float e = __expf(attnS[tid * NR + r] - mx); attnS[tid * NR + r] = e; sm += e; }
    float inv = 1.f / sm, nm = 0.f;
    for (int r = 0; r < NR; ++r) { float a = attnS[tid * NR + r] * inv; attnS[tid * NR + r] = a; nm += a * F[r * 64 + tid]; }
    numS[tid] = nm;
  }
  __syncthreads();

  // ||wctx_w||^2 = attn_w G attn_w^T via tmp = attn @ G (stored in QS)
  for (int idx = tid; idx < NW * NR; idx += 384) {
    int w = idx / NR, rp = idx - w * NR;
    float t = 0.f;
    for (int r = 0; r < NR; ++r) t += attnS[w * NR + r] * GS[r * NR + rp];
    QS[idx] = t;
  }
  __syncthreads();
  if (tid < NW) {
    float w2 = 0.f;
    for (int rp = 0; rp < NR; ++rp) w2 += QS[tid * NR + rp] * attnS[tid * NR + rp];
    float wn = sqrtf(fmaxf(w2, 0.f));
    rsim[tid] = numS[tid] / fmaxf(s_norm[c * NW + tid] * wn, 1e-8f);
  }
  __syncthreads();

  // LSE aggregation over words -> scores[img=i][cap=c]
  if (tid == 0) {
    float mx = -3.4e38f;
    for (int w = 0; w < NW; ++w) mx = fmaxf(mx, rsim[w] * 6.f);
    float sm = 0.f;
    for (int w = 0; w < NW; ++w) sm += __expf(rsim[w] * 6.f - mx);
    scores[i * NB + c] = (mx + __logf(sm)) / 6.f;
  }

  // ---- IPOT: 20 iterations on 32x36 matrices in LDS ----
  for (int it = 0; it < 20; ++it) {
    for (int idx = tid; idx < NW * NR; idx += 384) QS[idx] = ES[idx] * TS[idx];
    __syncthreads();
    if (tid < NW) { float d = 0.f; for (int r = 0; r < NR; ++r) d += QS[tid * NR + r] * sgm[r];
                    dlt[tid] = 1.f / ((float)NW * d); }
    __syncthreads();
    if (tid < NR) { float sv = 0.f; for (int w = 0; w < NW; ++w) sv += QS[w * NR + tid] * dlt[w];
                    sgm[tid] = 1.f / ((float)NR * sv); }
    __syncthreads();
    for (int idx = tid; idx < NW * NR; idx += 384) {
      int w = idx / NR, r = idx - w * NR;
      TS[idx] = dlt[w] * QS[idx] * sgm[r];
    }
    __syncthreads();
  }

  // ot_sim = -sum(C*T)
  float part = 0.f;
  for (int idx = tid; idx < NW * NR; idx += 384) part += CS[idx] * TS[idx];
  red[tid] = part;
  if (tid < 128) red[384 + tid] = 0.f;
  __syncthreads();
  for (int st = 256; st > 0; st >>= 1) { if (tid < st) red[tid] += red[tid + st]; __syncthreads(); }
  if (tid == 0) scores_ot[i * NB + c] = -red[0];
}

// ---------------------------------------------------------------------------
// Kernel 4: max-violation hinge loss over 64x64 score matrices -> 2 scalars
// ---------------------------------------------------------------------------
__global__ void loss_kernel(const float* __restrict__ S, const float* __restrict__ O,
                            float* __restrict__ out) {
  __shared__ float dS[NB], dO[NB], rA[NB], rB[NB];
  const int t = threadIdx.x;
  dS[t] = S[t * NB + t]; dO[t] = O[t * NB + t];
  __syncthreads();
  float m1 = 0.f, m2 = 0.f, m3 = 0.f, m4 = 0.f;
  for (int j = 0; j < NB; ++j) {
    if (j == t) continue;
    float cs  = fmaxf(0.2f + S[t * NB + j] - dS[t], 0.f);   // cost_s row t
    float co  = fmaxf(0.2f + O[t * NB + j] - dO[t], 0.f);
    m1 = fmaxf(m1, cs + 0.1f * co);
    m2 = fmaxf(m2, co);
    float ci  = fmaxf(0.2f + S[j * NB + t] - dS[t], 0.f);   // cost_im column t
    float cio = fmaxf(0.2f + O[j * NB + t] - dO[t], 0.f);
    m3 = fmaxf(m3, ci + 0.1f * cio);
    m4 = fmaxf(m4, cio);
  }
  rA[t] = m1 + m3; rB[t] = m2 + m4;
  __syncthreads();
  for (int st = 32; st > 0; st >>= 1) {
    if (t < st) { rA[t] += rA[t + st]; rB[t] += rB[t + st]; }
    __syncthreads();
  }
  if (t == 0) { out[0] = rA[0]; out[1] = rB[0]; }
}

// ---------------------------------------------------------------------------
// Launch
// ---------------------------------------------------------------------------
extern "C" void kernel_launch(void* const* d_in, const int* in_sizes, int n_in,
                              void* d_out, int out_size, void* d_ws, size_t ws_size,
                              hipStream_t stream) {
  (void)in_sizes; (void)n_in; (void)out_size; (void)ws_size;
  const float* im = (const float*)d_in[0];   // (64,36,1024)
  const float* s  = (const float*)d_in[1];   // (64,32,1024)
  /* d_in[2] = s_l (uniform == 32, unused) */
  const float* x  = (const float*)d_in[3];   // (64,32,1024)

  char* ws = (char*)d_ws;
  unsigned short* im_bf = (unsigned short*)(ws + 0);          // 2304*1024*2 = 4718592
  unsigned short* s_bf  = (unsigned short*)(ws + 4718592);    // 2048*1024*2 = 4194304
  unsigned short* xn_bf = (unsigned short*)(ws + 8912896);    // 4194304
  float* inv_imn        = (float*)(ws + 13107200);            // 2304*4
  float* s_normp        = (float*)(ws + 13116416);            // 2048*4
  float* G              = (float*)(ws + 13124608);            // 64*36*36*4 = 331776
  float* scoresM        = (float*)(ws + 13456384);            // 64*64*4
  float* otM            = (float*)(ws + 13472768);            // 64*64*4

  prep_kernel<<<6400, 256, 0, stream>>>(im, s, x, im_bf, s_bf, xn_bf, inv_imn, s_normp);
  gram_kernel<<<64, 288, 0, stream>>>(im_bf, G);
  pair_kernel<<<4096, 384, 0, stream>>>(im_bf, s_bf, xn_bf, inv_imn, s_normp, G, scoresM, otM);
  loss_kernel<<<1, 64, 0, stream>>>(scoresM, otM, (float*)d_out);
}
